// BaseColbertLoss_12077448036546
// MI455X (gfx1250) — matrix operand order repaired and checked by reading, not compiled, not run on hardware
//
#include <hip/hip_runtime.h>

// ---- CDNA5 WMMA vector types (bridge-doc verified) ----
typedef _Float16 half16 __attribute__((ext_vector_type(16)));
typedef float    float8 __attribute__((ext_vector_type(8)));

#define NQ   64     // n_queries
#define ND   64     // n_docs
#define TQ   32     // query tokens (2 M-tiles of 16 per query)
#define TD   1024   // doc tokens  (N total)
#define DIM  128    // K total = 4 blocks of 32
#define QPB  2      // queries per block (amortize B fragments over 4 M-tiles)

// Convert four float4 runs (two runs of 8 consecutive f32, 16 apart) into one f16 WMMA fragment.
// (16 elements = 8 VGPRs; element 2j/2j+1 = lo/hi half of VGPR j -> v_cvt_pk_f16_f32)
__device__ __forceinline__ half16 cvt16(float4 x0, float4 x1, float4 y0, float4 y1) {
    half16 a;
    a[0]  = (_Float16)x0.x; a[1]  = (_Float16)x0.y; a[2]  = (_Float16)x0.z; a[3]  = (_Float16)x0.w;
    a[4]  = (_Float16)x1.x; a[5]  = (_Float16)x1.y; a[6]  = (_Float16)x1.z; a[7]  = (_Float16)x1.w;
    a[8]  = (_Float16)y0.x; a[9]  = (_Float16)y0.y; a[10] = (_Float16)y0.z; a[11] = (_Float16)y0.w;
    a[12] = (_Float16)y1.x; a[13] = (_Float16)y1.y; a[14] = (_Float16)y1.z; a[15] = (_Float16)y1.w;
    return a;
}

__global__ __launch_bounds__(256)
void colbert_maxsim_wmma(const float* __restrict__ qs,
                         const float* __restrict__ ps,
                         float* __restrict__ out)
{
    const int c      = blockIdx.x;          // doc
    const int b0     = blockIdx.y * QPB;    // first query of this block
    const int tid    = threadIdx.x;
    const int wave   = tid >> 5;            // 0..7
    const int lane   = tid & 31;
    const int lane16 = lane & 15;
    const int hi     = lane >> 4;           // lane half selects K sub-block

    const float* __restrict__ pbase = ps + (size_t)c * TD * DIM;

    // ---------- Preload A fragments: 2 queries x 2 M-tiles x 4 K-blocks ----------
    // ISA 7.12.2, 16-bit A 16x32 (MxK):
    //   lanes 0-15 : M = lane,    K runs {kb*32 .. +7} and {kb*32+16 .. +23}
    //   lanes16-31 : M = lane-16, K runs {kb*32+8 .. +15} and {kb*32+24 .. +31}
    half16 afrag[QPB][2][4];
    const int khalf = hi * 8;
    #pragma unroll
    for (int q = 0; q < QPB; ++q) {
        const float* qb = qs + (size_t)(b0 + q) * TQ * DIM;
        #pragma unroll
        for (int mt = 0; mt < 2; ++mt) {
            const float* qrow = qb + (size_t)(mt * 16 + lane16) * DIM;
            #pragma unroll
            for (int kb = 0; kb < 4; ++kb) {
                const float4* r0 = (const float4*)(qrow + kb * 32 + khalf);
                const float4* r1 = (const float4*)(qrow + kb * 32 + 16 + khalf);
                afrag[q][mt][kb] = cvt16(r0[0], r0[1], r1[0], r1[1]);
            }
        }
    }

    // ---------- Stream B tiles (doc tokens); 4 independent WMMA chains per fragment ----------
    float8 rmax[QPB][2];
    #pragma unroll
    for (int q = 0; q < QPB; ++q)
        #pragma unroll
        for (int mt = 0; mt < 2; ++mt)
            #pragma unroll
            for (int i = 0; i < 8; ++i) rmax[q][mt][i] = -__builtin_inff();

    const int s_wave = wave * 128;          // this wave's 128 doc tokens
    #pragma unroll
    for (int t = 0; t < 8; ++t) {
        const int s0 = s_wave + t * 16;
        // B 32x16 (KxN): lane L<16 -> N = s0+L, K runs {kb*32 .. +7} and {kb*32+16 .. +23};
        //                lane L>=16 -> same N,  K runs {kb*32+8 .. +15} and {kb*32+24 .. +31}.
        const float* prow = pbase + (size_t)(s0 + lane16) * DIM + hi * 8;

        // Near prefetch of next tile's lines while this tile's WMMAs run.
        if (t < 7) __builtin_prefetch(prow + 16 * DIM, 0, 3);

        float8 acc[QPB][2] = {};
        #pragma unroll
        for (int kb = 0; kb < 4; ++kb) {
            const float4* pp = (const float4*)(prow + kb * 32);
            half16 bfrag = cvt16(pp[0], pp[1], pp[4], pp[5]);
            #pragma unroll
            for (int q = 0; q < QPB; ++q) {
                #pragma unroll
                for (int mt = 0; mt < 2; ++mt) {
                    acc[q][mt] = __builtin_amdgcn_wmma_f32_16x16x32_f16(
                        false, afrag[q][mt][kb], false, bfrag, (short)0, acc[q][mt],
                        false, false);
                }
            }
        }
        // Running max over doc-token tiles (elementwise in C/D layout).
        #pragma unroll
        for (int q = 0; q < QPB; ++q)
            #pragma unroll
            for (int mt = 0; mt < 2; ++mt)
                #pragma unroll
                for (int i = 0; i < 8; ++i)
                    rmax[q][mt][i] = fmaxf(rmax[q][mt][i], acc[q][mt][i]);
    }

    // ---------- Max across N (lanes within each 16-lane half) ----------
    // C/D layout: VGPR i holds row M=i (lanes 0-15, N=lane) and M=i+8 (lanes 16-31).
    #pragma unroll
    for (int off = 8; off >= 1; off >>= 1) {
        #pragma unroll
        for (int q = 0; q < QPB; ++q)
            #pragma unroll
            for (int mt = 0; mt < 2; ++mt)
                #pragma unroll
                for (int i = 0; i < 8; ++i)
                    rmax[q][mt][i] = fmaxf(rmax[q][mt][i],
                                           __shfl_xor(rmax[q][mt][i], off, 16));
    }

    // ---------- Cross-wave max + final sum via LDS ----------
    __shared__ float wmax[8][QPB][TQ];      // [wave][query][query-token row]
    if (lane == 0) {
        #pragma unroll
        for (int q = 0; q < QPB; ++q)
            #pragma unroll
            for (int i = 0; i < 8; ++i) {
                wmax[wave][q][i]      = rmax[q][0][i];  // rows 0..7
                wmax[wave][q][16 + i] = rmax[q][1][i];  // rows 16..23
            }
    } else if (lane == 16) {
        #pragma unroll
        for (int q = 0; q < QPB; ++q)
            #pragma unroll
            for (int i = 0; i < 8; ++i) {
                wmax[wave][q][8 + i]  = rmax[q][0][i];  // rows 8..15
                wmax[wave][q][24 + i] = rmax[q][1][i];  // rows 24..31
            }
    }
    __syncthreads();

    if (wave < QPB) {                        // wave q reduces query b0+q
        float m = wmax[0][wave][lane];       // lane = query-token row (0..31)
        #pragma unroll
        for (int w = 1; w < 8; ++w) m = fmaxf(m, wmax[w][wave][lane]);
        #pragma unroll
        for (int off = 16; off >= 1; off >>= 1) m += __shfl_xor(m, off, 32);
        if (lane == 0) out[(size_t)(b0 + wave) * ND + c] = m;
    }
}

extern "C" void kernel_launch(void* const* d_in, const int* in_sizes, int n_in,
                              void* d_out, int out_size, void* d_ws, size_t ws_size,
                              hipStream_t stream) {
    (void)in_sizes; (void)n_in; (void)out_size; (void)d_ws; (void)ws_size;
    const float* qs = (const float*)d_in[0];   // (64, 32, 128) f32
    const float* ps = (const float*)d_in[1];   // (64, 1024, 128) f32
    float* out = (float*)d_out;                // (64, 64) f32

    dim3 grid(ND, NQ / QPB);                   // one WG per (doc, query-pair)
    dim3 block(256);                           // 8 waves of 32
    colbert_maxsim_wmma<<<grid, block, 0, stream>>>(qs, ps, out);
}
